// DecoderLayer_28827820490793
// MI455X (gfx1250) — matrix operand order repaired and checked by reading, not compiled, not run on hardware
//
#include <hip/hip_runtime.h>

// ---------------- problem constants ----------------
constexpr int kD  = 256, kH = 8, kB = 16, kNP = 128, kEP = 256, kS = 1024;
constexpr int kN  = kB * kNP;      // 2048 nodes
constexpr int kE  = kB * kEP;      // 4096 edges
constexpr int kL  = kNP + kEP;     // 384 tokens per batch
constexpr int kDH = kD / kH;       // 32
constexpr int kNE = kN + kE;       // 6144 graph rows
constexpr int kBL = kB * kL;       // 6144 (== kNE)
constexpr int kBS = kB * kS;       // 16384
constexpr int kD4 = 4 * kD;        // 1024

// ---------------- WMMA types / helpers ----------------
typedef __bf16 bf16;
typedef __attribute__((ext_vector_type(16))) __bf16 v16bf;
typedef __attribute__((ext_vector_type(8)))  __bf16 v8bf;
typedef __attribute__((ext_vector_type(8)))  float  v8f;

__device__ inline v8f wmma_bf16(v16bf a, v16bf b, v8f c) {
  // D = A(16x32 bf16) * B(32x16 bf16) + C(16x16 f32)
  return __builtin_amdgcn_wmma_f32_16x16x32_bf16(
      false, a, false, b, (short)0, c, false, false);
}

// A-fragment (16x32 bf16): lane lo holds row M=lo; lo lanes K{0..15},
// hi lanes K{16..31}  -> two contiguous 16B loads.
__device__ inline v16bf load_a_frag(const bf16* __restrict__ rowp, int hi) {
  v8bf x0 = *(const v8bf*)(rowp + hi * 16);
  v8bf x1 = *(const v8bf*)(rowp + 8 + hi * 16);
  v16bf r;
#pragma unroll
  for (int j = 0; j < 8; ++j) { r[j] = x0[j]; r[j + 8] = x1[j]; }
  return r;
}

__device__ inline v16bf load_a_frag_lds(const bf16* rowp, int hi) {
  v8bf x0 = *(const v8bf*)(rowp + hi * 16);
  v8bf x1 = *(const v8bf*)(rowp + 8 + hi * 16);
  v16bf r;
#pragma unroll
  for (int j = 0; j < 8; ++j) { r[j] = x0[j]; r[j + 8] = x1[j]; }
  return r;
}

__device__ inline float gelu_tanh(float x) {
  float x3 = x * x * x;
  return 0.5f * x * (1.f + tanhf(0.7978845608028654f * (x + 0.044715f * x3)));
}

__device__ inline int graph_to_bl(int r) {
  if (r < kN) { int b = r / kNP; return b * kL + (r - b * kNP); }
  int rr = r - kN; int b = rr / kEP; return b * kL + kNP + (rr - b * kEP);
}

// ---------------- small utility kernels ----------------
__global__ void fill_f32(float* p, float v, int n) {
  int i = blockIdx.x * 256 + threadIdx.x; if (i < n) p[i] = v;
}
__global__ void fill_i32(int* p, int v, int n) {
  int i = blockIdx.x * 256 + threadIdx.x; if (i < n) p[i] = v;
}
__global__ void cast_bf16_kernel(const float* __restrict__ in, bf16* __restrict__ out, int n) {
  int i = blockIdx.x * 256 + threadIdx.x; if (i < n) out[i] = (bf16)in[i];
}

// Pack weight W[Nout,K] (row-major fp32) into WMMA B fragment-major bf16:
// P[((kt*ntiles+nt)*32 + lane)*16 + j] = W[nt*16+(lane&15)][kt*32+(lane>>4)*16+j]
__global__ void pack_wT(const float* __restrict__ W, bf16* __restrict__ P, int Nout, int K) {
  int idx = blockIdx.x * 256 + threadIdx.x;
  if (idx >= Nout * K) return;
  int j = idx & 15, lane = (idx >> 4) & 31, t = idx >> 9;
  int ntiles = Nout >> 4;
  int nt = t % ntiles, kt = t / ntiles;
  int n = nt * 16 + (lane & 15);
  int k = kt * 32 + (lane >> 4) * 16 + j;
  P[idx] = (bf16)W[(size_t)n * K + k];
}

// Pack K-proj [BS,D] bf16 into attention score B-fragments per (b,h):
// tile nt over S/16, k = dh (one 32-ktile)
__global__ void pack_attnK(const bf16* __restrict__ Kp, bf16* __restrict__ P) {
  int idx = blockIdx.x * 256 + threadIdx.x;
  if (idx >= kB * kH * kS * kDH) return;
  int j = idx & 15, lane = (idx >> 4) & 31, t = idx >> 9;
  int st = t % (kS / 16); int bh = t / (kS / 16);
  int b = bh / kH, h = bh % kH;
  int s = st * 16 + (lane & 15);
  int dh = (lane >> 4) * 16 + j;
  P[idx] = Kp[((size_t)b * kS + s) * kD + h * kDH + dh];
}

// Pack V-proj [BS,D] bf16 into ctx B-fragments per (b,h):
// tiles: (S/32) ktiles x 2 ntiles (dh halves); B(k=s_local, n=dh)
__global__ void pack_attnV(const bf16* __restrict__ Vp, bf16* __restrict__ P) {
  int idx = blockIdx.x * 256 + threadIdx.x;
  if (idx >= kB * kH * kS * kDH) return;
  int j = idx & 15, lane = (idx >> 4) & 31, t = idx >> 9;
  int nt2 = t & 1; int st = (t >> 1) % (kS / 32); int bh = t / (2 * (kS / 32));
  int b = bh / kH, h = bh % kH;
  int s = st * 32 + (lane >> 4) * 16 + j;
  int dh = nt2 * 16 + (lane & 15);
  P[idx] = Vp[((size_t)b * kS + s) * kD + h * kDH + dh];
}

// ---------------- LayerNorm (one 256-thread block per row, D=256) ----------------
__global__ __launch_bounds__(256) void ln_kernel(
    const float* __restrict__ Xa, const float* __restrict__ Xb, int split,
    const float* __restrict__ g, const float* __restrict__ be,
    const float* __restrict__ embA, const float* __restrict__ embB,
    float* __restrict__ Yf, bf16* __restrict__ Yb, int remap)
{
  int row = blockIdx.x, tid = threadIdx.x;
  const float* x = (row < split) ? (Xa + (size_t)row * kD)
                                 : (Xb + (size_t)(row - split) * kD);
  float v = x[tid];
  __shared__ float s1[8], s2[8], st[2];
  float a = v, q = v * v;
#pragma unroll
  for (int off = 16; off; off >>= 1) {
    a += __shfl_xor(a, off, 32);
    q += __shfl_xor(q, off, 32);
  }
  if ((tid & 31) == 0) { s1[tid >> 5] = a; s2[tid >> 5] = q; }
  __syncthreads();
  if (tid == 0) {
    float sa = 0.f, sq = 0.f;
    for (int i = 0; i < 8; ++i) { sa += s1[i]; sq += s2[i]; }
    float mean = sa / kD;
    float var = sq / kD - mean * mean;
    st[0] = mean; st[1] = rsqrtf(var + 1e-5f);
  }
  __syncthreads();
  float y = (v - st[0]) * st[1] * g[tid] + be[tid];
  if (Yf) Yf[(size_t)row * kD + tid] = y;
  if (Yb) {
    float ye = y;
    if (embA) ye += (row < split) ? embA[(size_t)row * kD + tid]
                                  : embB[(size_t)(row - split) * kD + tid];
    int orow = remap ? graph_to_bl(row) : row;
    Yb[(size_t)orow * kD + tid] = (bf16)ye;
  }
}

// ---------------- generic WMMA GEMM: C[M,Nc] = A[M,K] @ W^T (+bias) ----------------
// A: bf16 row-major [M,K]. Bp: fragment-packed bf16 (see pack_wT).
// Block 128 threads (4 waves), 64x64 tile; wave = 32x32 via 2x2 WMMA tiles.
// EPI: 0 = none, 1 = +bias, 2 = +bias + gelu(tanh)
template <int EPI>
__global__ __launch_bounds__(128) void gemm_kernel(
    const bf16* __restrict__ A, const bf16* __restrict__ Bp,
    const float* __restrict__ bias,
    float* __restrict__ Cf, bf16* __restrict__ Cb,
    int M, int Nc, int K)
{
  int lane = threadIdx.x & 31, wv = threadIdx.x >> 5;
  int lo = lane & 15, hi = lane >> 4;
  int bm = blockIdx.y * 64 + (wv >> 1) * 32;
  int bn = blockIdx.x * 64 + (wv & 1) * 32;
  int ntiles = Nc >> 4;
  v8f acc[2][2] = {};
  const bf16* Arow0 = A + (size_t)(bm + lo) * K;
  const bf16* Arow1 = A + (size_t)(bm + 16 + lo) * K;
  for (int k0 = 0; k0 < K; k0 += 32) {
    if (k0 + 32 < K) __builtin_prefetch(Arow0 + k0 + 32, 0, 3);
    v16bf a0 = load_a_frag(Arow0 + k0, hi);
    v16bf a1 = load_a_frag(Arow1 + k0, hi);
    size_t tb = (size_t)(k0 >> 5) * ntiles;
    v16bf b0 = *(const v16bf*)(Bp + ((tb + (bn >> 4)) * 32 + lane) * 16);
    v16bf b1 = *(const v16bf*)(Bp + ((tb + (bn >> 4) + 1) * 32 + lane) * 16);
    acc[0][0] = wmma_bf16(a0, b0, acc[0][0]);
    acc[0][1] = wmma_bf16(a0, b1, acc[0][1]);
    acc[1][0] = wmma_bf16(a1, b0, acc[1][0]);
    acc[1][1] = wmma_bf16(a1, b1, acc[1][1]);
  }
#pragma unroll
  for (int i = 0; i < 2; ++i)
#pragma unroll
    for (int j = 0; j < 2; ++j) {
      int col = bn + j * 16 + lo;
      float bb = (EPI >= 1 && bias) ? bias[col] : 0.f;
#pragma unroll
      for (int p = 0; p < 8; ++p) {
        int row = bm + i * 16 + p + 8 * hi;
        float y = acc[i][j][p] + bb;
        if (EPI == 2) y = gelu_tanh(y);
        if (Cf) Cf[(size_t)row * Nc + col] = y;
        if (Cb) Cb[(size_t)row * Nc + col] = (bf16)y;
      }
    }
}

// ---------------- fused flash attention per (b,h,16-row L-tile) ----------------
__global__ __launch_bounds__(32) void attn_kernel(
    const bf16* __restrict__ Qp,    // [BL, D]
    const bf16* __restrict__ Kpk,   // packed score-B [B*H][S/16][32][16]
    const bf16* __restrict__ Vpk,   // packed ctx-B   [B*H][S/32][2][32][16]
    const float* __restrict__ mask, // [B*H, L, S]
    bf16* __restrict__ Ctx)         // [BL, D]
{
  __shared__ bf16 p_lds[16][32];
  int lane = threadIdx.x, lo = lane & 15, hi = lane >> 4;
  int lt = blockIdx.x, bh = blockIdx.y;
  int b = bh / kH, h = bh % kH;
  int row0 = lt * 16;
  const float scale = 0.17677669529663687f; // 1/sqrt(32)

  v16bf aq = load_a_frag(Qp + ((size_t)b * kL + row0 + lo) * kD + h * kDH, hi);

  v8f ctx0 = {}, ctx1 = {};
  float m[8], ssum[8];
#pragma unroll
  for (int i = 0; i < 8; ++i) { m[i] = -1e30f; ssum[i] = 0.f; }

  for (int s0 = 0; s0 < kS; s0 += 32) {
    v8f z = {};
    v16bf bk0 = *(const v16bf*)(Kpk + (((size_t)bh * (kS / 16) + (s0 >> 4)) * 32 + lane) * 16);
    v16bf bk1 = *(const v16bf*)(Kpk + (((size_t)bh * (kS / 16) + (s0 >> 4) + 1) * 32 + lane) * 16);
    v8f sc0 = wmma_bf16(aq, bk0, z);
    v8f sc1 = wmma_bf16(aq, bk1, z);

    float nm[8];
#pragma unroll
    for (int i = 0; i < 8; ++i) {
      int r = row0 + i + 8 * hi;
      const float* mrow = mask + ((size_t)bh * kL + r) * kS + s0;
      float x0 = sc0[i] * scale + mrow[lo];
      float x1 = sc1[i] * scale + mrow[16 + lo];
      sc0[i] = x0; sc1[i] = x1;
      nm[i] = fmaxf(m[i], fmaxf(x0, x1));
    }
#pragma unroll
    for (int off = 1; off < 16; off <<= 1) {
#pragma unroll
      for (int i = 0; i < 8; ++i) nm[i] = fmaxf(nm[i], __shfl_xor(nm[i], off, 32));
    }
#pragma unroll
    for (int i = 0; i < 8; ++i) {
      float corr = __expf(m[i] - nm[i]);
      ctx0[i] *= corr; ctx1[i] *= corr;
      float p0 = __expf(sc0[i] - nm[i]);
      float p1 = __expf(sc1[i] - nm[i]);
      ssum[i] = ssum[i] * corr + p0 + p1;   // per-lane partial; reduced at end
      m[i] = nm[i];
      p_lds[i + 8 * hi][lo]      = (bf16)p0;
      p_lds[i + 8 * hi][16 + lo] = (bf16)p1;
    }
    __syncthreads();
    v16bf ap = load_a_frag_lds(&p_lds[lo][0], hi);
    size_t vb = (((size_t)bh * (kS / 32) + (s0 >> 5)) * 2);
    v16bf bv0 = *(const v16bf*)(Vpk + ((vb + 0) * 32 + lane) * 16);
    v16bf bv1 = *(const v16bf*)(Vpk + ((vb + 1) * 32 + lane) * 16);
    ctx0 = wmma_bf16(ap, bv0, ctx0);
    ctx1 = wmma_bf16(ap, bv1, ctx1);
    __syncthreads();
  }
#pragma unroll
  for (int off = 1; off < 16; off <<= 1) {
#pragma unroll
    for (int i = 0; i < 8; ++i) ssum[i] += __shfl_xor(ssum[i], off, 32);
  }
#pragma unroll
  for (int i = 0; i < 8; ++i) {
    float inv = 1.f / ssum[i];
    size_t r = (size_t)b * kL + row0 + i + 8 * hi;
    Ctx[r * kD + h * kDH + lo]      = (bf16)(ctx0[i] * inv);
    Ctx[r * kD + h * kDH + 16 + lo] = (bf16)(ctx1[i] * inv);
  }
}

// ---------------- residual kernels ----------------
__global__ void res1_kernel(const float* __restrict__ nodes, const float* __restrict__ edges,
                            const float* __restrict__ ctx, const float* __restrict__ ls1,
                            float* __restrict__ q1) {
  int idx = blockIdx.x * 256 + threadIdx.x;
  int r = idx >> 8, c = idx & 255;
  float base = (r < kN) ? nodes[idx] : edges[idx - kN * kD];
  int br = graph_to_bl(r);
  q1[idx] = base + ls1[c] * ctx[(size_t)br * kD + c];
}

__global__ void res2_kernel(const float* __restrict__ q2, const float* __restrict__ agg,
                            const float* __restrict__ epf, const float* __restrict__ gat_b,
                            const float* __restrict__ ls2, float* __restrict__ q3) {
  int idx = blockIdx.x * 256 + threadIdx.x;
  int r = idx >> 8, c = idx & 255;
  float g = (r < kN) ? (agg[idx] + gat_b[c]) : epf[idx - kN * kD];
  q3[idx] = q2[idx] + ls2[c] * g;
}

__global__ void res3_kernel(const float* __restrict__ q3, const float* __restrict__ ff,
                            const float* __restrict__ ls3, float* __restrict__ out) {
  int idx = blockIdx.x * 256 + threadIdx.x;
  int c = idx & 255;
  out[idx] = q3[idx] + ls3[c] * ff[idx];
}

// ---------------- GAT kernels ----------------
__device__ inline float key_to_float(int k) {
  return __int_as_float(k >= 0 ? k : (k ^ 0x7fffffff));
}

__global__ __launch_bounds__(256) void gat_logit_kernel(
    const float* __restrict__ x, const float* __restrict__ ep,
    const int* __restrict__ ei, const float* __restrict__ a_src,
    const float* __restrict__ a_dst, const float* __restrict__ a_edge,
    float* __restrict__ logit, int* __restrict__ mkey) {
  int e = blockIdx.x, t = threadIdx.x;          // t = h*32 + dh
  int src = ei[e], dst = ei[kE + e];
  float v = x[(size_t)src * kD + t] * a_src[t]
          + x[(size_t)dst * kD + t] * a_dst[t]
          + ep[(size_t)e * kD + t] * a_edge[t];
#pragma unroll
  for (int off = 16; off; off >>= 1) v += __shfl_xor(v, off, 32);
  if ((t & 31) == 0) {
    int h = t >> 5;
    float l = v > 0.f ? v : 0.2f * v;           // leaky_relu(0.2)
    logit[(size_t)e * kH + h] = l;
    int key = __float_as_int(l);
    key = key >= 0 ? key : (key ^ 0x7fffffff);
    atomicMax(&mkey[dst * kH + h], key);
  }
}

__global__ void gat_expden_kernel(const float* __restrict__ logit, const int* __restrict__ ei,
                                  const int* __restrict__ mkey, float* __restrict__ ex,
                                  float* __restrict__ den) {
  int idx = blockIdx.x * 256 + threadIdx.x;
  if (idx >= kE * kH) return;
  int e = idx / kH, h = idx % kH;
  int dst = ei[kE + e];
  float mf = key_to_float(mkey[dst * kH + h]);
  if (!__builtin_isfinite(mf)) mf = 0.f;
  float v = __expf(logit[idx] - mf);
  ex[idx] = v;
  atomicAdd(&den[dst * kH + h], v);
}

__global__ __launch_bounds__(256) void gat_agg_kernel(
    const float* __restrict__ x, const int* __restrict__ ei,
    const float* __restrict__ ex, const float* __restrict__ den,
    float* __restrict__ agg) {
  int e = blockIdx.x, t = threadIdx.x;
  int h = t >> 5;
  int src = ei[e], dst = ei[kE + e];
  float alpha = ex[(size_t)e * kH + h] / (den[dst * kH + h] + 1e-16f);
  atomicAdd(&agg[(size_t)dst * kD + t], alpha * x[(size_t)src * kD + t]);
}

// ---------------- host launcher ----------------
extern "C" void kernel_launch(void* const* d_in, const int* in_sizes, int n_in,
                              void* d_out, int out_size, void* d_ws, size_t ws_size,
                              hipStream_t stream) {
  (void)in_sizes; (void)n_in; (void)out_size; (void)ws_size;
  const float* nodes      = (const float*)d_in[0];
  const float* edges      = (const float*)d_in[1];
  const float* features   = (const float*)d_in[2];
  const float* attn_mask  = (const float*)d_in[3];
  const float* emb_nodes  = (const float*)d_in[4];
  const float* emb_edges  = (const float*)d_in[5];
  const int*   edge_index = (const int*)d_in[6];
  const float* ln1_g = (const float*)d_in[7];
  const float* ln1_b = (const float*)d_in[8];
  const float* w_qkv = (const float*)d_in[9];
  const float* b_qkv = (const float*)d_in[10];
  const float* w_o   = (const float*)d_in[11];
  const float* b_o   = (const float*)d_in[12];
  const float* ls1   = (const float*)d_in[13];
  const float* ln2_g = (const float*)d_in[14];
  const float* ln2_b = (const float*)d_in[15];
  const float* w_n   = (const float*)d_in[16];
  const float* w_e   = (const float*)d_in[17];
  const float* a_src = (const float*)d_in[18];
  const float* a_dst = (const float*)d_in[19];
  const float* a_edge= (const float*)d_in[20];
  const float* gat_b = (const float*)d_in[21];
  const float* ls2   = (const float*)d_in[22];
  const float* ln3_g = (const float*)d_in[23];
  const float* ln3_b = (const float*)d_in[24];
  const float* w1    = (const float*)d_in[25];
  const float* b1    = (const float*)d_in[26];
  const float* w2    = (const float*)d_in[27];
  const float* b2    = (const float*)d_in[28];
  const float* ls3   = (const float*)d_in[29];
  float* out = (float*)d_out;

  // ---- workspace layout (lifetime-based reuse) ----
  char* w = (char*)d_ws;
  size_t o = 0;
  auto take = [&](size_t bytes) { size_t r = o; o += (bytes + 255) & ~(size_t)255; return r; };
  size_t o_wq  = take((size_t)kD * kD * 2);
  size_t o_wk  = take((size_t)kD * kD * 2);
  size_t o_wv  = take((size_t)kD * kD * 2);
  size_t o_wo  = take((size_t)kD * kD * 2);
  size_t o_wn  = take((size_t)kD * kD * 2);
  size_t o_we  = take((size_t)kD * kD * 2);
  size_t o_w1  = take((size_t)kD * kD4 * 2);
  size_t o_w2  = take((size_t)kD4 * kD * 2);
  size_t o_qbf = take((size_t)kBL * kD * 2);        // q1 LN bf16; reused as q4 bf16
  size_t o_feat= take((size_t)kBS * kD * 2);        // features bf16; reused as ctx_o f32
  size_t o_Qp  = take((size_t)kBL * kD * 2);        // Q proj bf16; (Qp..Ctx) reused as ffn f32
  size_t o_Ctx = take((size_t)kBL * kD * 2);        // attention ctx bf16
  size_t o_R   = take((size_t)2 * kBS * kD * 2);    // Kp|Vp bf16 -> q1 f32 -> h1 bf16
  size_t o_Kpk = take((size_t)kBS * kD * 2);        // packed K -> xf f32 + epf f32
  size_t o_Vpk = take((size_t)kBS * kD * 2);        // packed V -> q2 f32
  size_t o_hbf = take((size_t)kNE * kD * 2);        // hn/he bf16
  size_t o_q3  = take((size_t)kNE * kD * 4);
  size_t o_logit = take((size_t)kE * kH * 4);
  size_t o_ex    = take((size_t)kE * kH * 4);
  size_t o_mkey  = take((size_t)kN * kH * 4);
  size_t o_den   = take((size_t)kN * kH * 4);
  size_t o_agg   = take((size_t)kN * kD * 4);

  bf16* wqP = (bf16*)(w + o_wq); bf16* wkP = (bf16*)(w + o_wk);
  bf16* wvP = (bf16*)(w + o_wv); bf16* woP = (bf16*)(w + o_wo);
  bf16* wnP = (bf16*)(w + o_wn); bf16* weP = (bf16*)(w + o_we);
  bf16* w1P = (bf16*)(w + o_w1); bf16* w2P = (bf16*)(w + o_w2);
  bf16* qbf  = (bf16*)(w + o_qbf);
  bf16* featb= (bf16*)(w + o_feat);
  bf16* Qp   = (bf16*)(w + o_Qp);
  bf16* Ctxb = (bf16*)(w + o_Ctx);
  bf16* Kp   = (bf16*)(w + o_R);
  bf16* Vp   = (bf16*)(w + o_R + (size_t)kBS * kD * 2);
  bf16* Kpk  = (bf16*)(w + o_Kpk);
  bf16* Vpk  = (bf16*)(w + o_Vpk);
  bf16* hbf  = (bf16*)(w + o_hbf);
  float* ctx_o = (float*)(w + o_feat);              // reuse featb region
  float* q1    = (float*)(w + o_R);                 // reuse Kp region
  float* q2    = (float*)(w + o_Vpk);               // reuse Vpk region
  float* xf    = (float*)(w + o_Kpk);               // reuse Kpk region
  float* epf   = (float*)(w + o_Kpk + (size_t)kN * kD * 4);
  float* q3    = (float*)(w + o_q3);
  bf16*  q4bf  = (bf16*)(w + o_qbf);                // reuse qbf region
  bf16*  h1bf  = (bf16*)(w + o_R);                  // reuse Kp|Vp region (12.6MB <= 16.8MB)
  float* fff   = (float*)(w + o_Qp);                // reuse Qp|Ctx region (6.29MB)
  float* logit = (float*)(w + o_logit);
  float* ex    = (float*)(w + o_ex);
  int*   mkey  = (int*)(w + o_mkey);
  float* den   = (float*)(w + o_den);
  float* agg   = (float*)(w + o_agg);

  auto cdiv = [](int a, int b) { return (a + b - 1) / b; };

  // ---- 1. pack weights (fp32 -> fragment-major bf16) ----
  pack_wT<<<cdiv(kD * kD, 256), 256, 0, stream>>>(w_qkv,              wqP, kD,  kD);
  pack_wT<<<cdiv(kD * kD, 256), 256, 0, stream>>>(w_qkv + kD * kD,    wkP, kD,  kD);
  pack_wT<<<cdiv(kD * kD, 256), 256, 0, stream>>>(w_qkv + 2 * kD * kD,wvP, kD,  kD);
  pack_wT<<<cdiv(kD * kD, 256), 256, 0, stream>>>(w_o,                woP, kD,  kD);
  pack_wT<<<cdiv(kD * kD, 256), 256, 0, stream>>>(w_n,                wnP, kD,  kD);
  pack_wT<<<cdiv(kD * kD, 256), 256, 0, stream>>>(w_e,                weP, kD,  kD);
  pack_wT<<<cdiv(kD4 * kD, 256), 256, 0, stream>>>(w1,                w1P, kD4, kD);
  pack_wT<<<cdiv(kD * kD4, 256), 256, 0, stream>>>(w2,                w2P, kD,  kD4);

  // ---- 2. LN1 over [nodes;edges] -> bf16 in Q (B*L) order ----
  ln_kernel<<<kNE, 256, 0, stream>>>(nodes, edges, kN, ln1_g, ln1_b,
                                     nullptr, nullptr, nullptr, qbf, 1);

  // ---- 3. Q projection ----
  gemm_kernel<1><<<dim3(kD / 64, kBL / 64), 128, 0, stream>>>(
      qbf, wqP, b_qkv, nullptr, Qp, kBL, kD, kD);

  // ---- 4-5. K/V projections from features ----
  cast_bf16_kernel<<<cdiv(kBS * kD, 256), 256, 0, stream>>>(features, featb, kBS * kD);
  gemm_kernel<1><<<dim3(kD / 64, kBS / 64), 128, 0, stream>>>(
      featb, wkP, b_qkv + kD, nullptr, Kp, kBS, kD, kD);
  gemm_kernel<1><<<dim3(kD / 64, kBS / 64), 128, 0, stream>>>(
      featb, wvP, b_qkv + 2 * kD, nullptr, Vp, kBS, kD, kD);

  // ---- 6. pack K/V into WMMA B-fragment layouts ----
  pack_attnK<<<cdiv(kB * kH * kS * kDH, 256), 256, 0, stream>>>(Kp, Kpk);
  pack_attnV<<<cdiv(kB * kH * kS * kDH, 256), 256, 0, stream>>>(Vp, Vpk);

  // ---- 7. fused flash attention ----
  attn_kernel<<<dim3(kL / 16, kB * kH), 32, 0, stream>>>(Qp, Kpk, Vpk, attn_mask, Ctxb);

  // ---- 8. output projection ----
  gemm_kernel<1><<<dim3(kD / 64, kBL / 64), 128, 0, stream>>>(
      Ctxb, woP, b_o, ctx_o, nullptr, kBL, kD, kD);

  // ---- 9. residual 1 (remap BL -> graph order) ----
  res1_kernel<<<cdiv(kNE * kD, 256), 256, 0, stream>>>(nodes, edges, ctx_o, ls1, q1);

  // ---- 10. LN2 -> q2 f32 and (q2+emb) bf16 ----
  ln_kernel<<<kNE, 256, 0, stream>>>(q1, q1 + (size_t)kN * kD, kN, ln2_g, ln2_b,
                                     emb_nodes, emb_edges, q2, hbf, 0);

  // ---- 11. GAT projections ----
  gemm_kernel<0><<<dim3(kD / 64, kN / 64), 128, 0, stream>>>(
      hbf, wnP, nullptr, xf, nullptr, kN, kD, kD);
  gemm_kernel<0><<<dim3(kD / 64, kE / 64), 128, 0, stream>>>(
      hbf + (size_t)kN * kD, weP, nullptr, epf, nullptr, kE, kD, kD);

  // ---- 12-15. GAT segment softmax + aggregation ----
  fill_i32<<<cdiv(kN * kH, 256), 256, 0, stream>>>(mkey, (int)0x80000000, kN * kH);
  fill_f32<<<cdiv(kN * kH, 256), 256, 0, stream>>>(den, 0.f, kN * kH);
  fill_f32<<<cdiv(kN * kD, 256), 256, 0, stream>>>(agg, 0.f, kN * kD);
  gat_logit_kernel<<<kE, 256, 0, stream>>>(xf, epf, edge_index, a_src, a_dst, a_edge,
                                           logit, mkey);
  gat_expden_kernel<<<cdiv(kE * kH, 256), 256, 0, stream>>>(logit, edge_index, mkey, ex, den);
  gat_agg_kernel<<<kE, 256, 0, stream>>>(xf, edge_index, ex, den, agg);

  // ---- 16. residual 2 ----
  res2_kernel<<<cdiv(kNE * kD, 256), 256, 0, stream>>>(q2, agg, epf, gat_b, ls2, q3);

  // ---- 17. LN3 -> q4 bf16 ----
  ln_kernel<<<kNE, 256, 0, stream>>>(q3, q3 + (size_t)kN * kD, kN, ln3_g, ln3_b,
                                     nullptr, nullptr, nullptr, q4bf, 0);

  // ---- 18-19. FFN ----
  gemm_kernel<2><<<dim3(kD4 / 64, kNE / 64), 128, 0, stream>>>(
      q4bf, w1P, b1, nullptr, h1bf, kNE, kD4, kD);
  gemm_kernel<1><<<dim3(kD / 64, kNE / 64), 128, 0, stream>>>(
      h1bf, w2P, b2, fff, nullptr, kNE, kD, kD4);

  // ---- 20. final residual -> out ----
  res3_kernel<<<cdiv(kNE * kD, 256), 256, 0, stream>>>(q3, fff, ls3, out);
}
